// TransformerBlock_72327249264824
// MI455X (gfx1250) — compile-verified, matrix operand order checked by reading, not run on hardware
//
#include <hip/hip_runtime.h>
#include <stdint.h>

#define B_  2
#define S_  2048
#define D_  1024
#define H_  16
#define FF_ 4096
#define HD_ 64
#define BS_ (B_*S_)   // 4096 rows

typedef __attribute__((ext_vector_type(16))) __bf16 bf16x16;
typedef __attribute__((ext_vector_type(8)))  __bf16 bf16x8;
typedef __attribute__((ext_vector_type(8)))  float  f32x8;
typedef __attribute__((ext_vector_type(4)))  unsigned int u32x4;
typedef __attribute__((ext_vector_type(8)))  int i32x8;
typedef __attribute__((ext_vector_type(4)))  int i32x4;

#if __has_builtin(__builtin_amdgcn_tensor_load_to_lds)
#define HAVE_TDM 1
#else
#define HAVE_TDM 0
#endif

// ---------- bf16 helpers (bit-exact storage via uint16) ----------
__device__ __forceinline__ uint16_t f2bf(float f) {
    union { float f; uint32_t u; } v; v.f = f;
    uint32_t u = v.u;
    uint32_t r = u + 0x7FFFu + ((u >> 16) & 1u);       // round-nearest-even
    if ((u & 0x7F800000u) == 0x7F800000u) r = u;       // inf/nan passthrough
    return (uint16_t)(r >> 16);
}
__device__ __forceinline__ float bf2f(uint16_t h) {
    union { uint32_t u; float f; } v; v.u = ((uint32_t)h) << 16;
    return v.f;
}

// ---------- WMMA fragment load from LDS ----------
// A/B 16-bit fragment layout (CDNA5 ISA 7.12.2): lane L holds row (L&15);
// lanes 0-15: elems 0..7 = K 0..7, elems 8..15 = K 16..23
// lanes 16-31: elems 0..7 = K 8..15, elems 8..15 = K 24..31
__device__ __forceinline__ bf16x16 load_frag_lds(const uint16_t* rowbase, int kh8) {
    bf16x8 lo = *(const bf16x8*)(rowbase + kh8);
    bf16x8 hi = *(const bf16x8*)(rowbase + 16 + kh8);
    return __builtin_shufflevector(lo, hi, 0,1,2,3,4,5,6,7,8,9,10,11,12,13,14,15);
}

__device__ __forceinline__ f32x8 wmma_bf16(bf16x16 a, bf16x16 b, f32x8 c) {
    return __builtin_amdgcn_wmma_f32_16x16x32_bf16(false, a, false, b, (short)0, c, false, false);
}

// ---------- Tensor Data Mover: 2D bf16 tile global->LDS ----------
// D# packing per CDNA5 ISA 8.3/8.4. pad: 4 dwords after every 16 dwords
// => LDS row pitch = 32 elems + 8 pad = 40 (matches LPg below).
#if HAVE_TDM
__device__ __forceinline__ void tdm_load_tile_2d(
    uint32_t lds_off, const uint16_t* gsrc,
    uint32_t tensor_d0, uint32_t tensor_d1,
    uint32_t tile_d0, uint32_t tile_d1, uint64_t stride0) {
    const uint64_t ga = (uint64_t)(size_t)gsrc;
    u32x4 g0;
    g0[0] = 1u;                                          // count=1, no gather
    g0[1] = lds_off;                                     // lds_addr (bytes)
    g0[2] = (uint32_t)(ga & 0xFFFFFFFFu);                // global_addr[31:0]
    g0[3] = (uint32_t)((ga >> 32) & 0x01FFFFFFu) | (2u << 30); // addr[56:32] | type=2
    i32x8 g1;
    // workgroup_mask=0 | data_size=1 (2B) | pad_enable | pad_interval=3 (16dw) | pad_amount=3 (4dw)
    g1[0] = (int)((1u << 16) | (1u << 20) | (3u << 22) | (3u << 25));
    g1[1] = (int)((tensor_d0 & 0xFFFFu) << 16);                          // dim0[15:0]
    g1[2] = (int)((tensor_d0 >> 16) | ((tensor_d1 & 0xFFFFu) << 16));    // dim0[31:16] | dim1[15:0]
    g1[3] = (int)((tensor_d1 >> 16) | (tile_d0 << 16));                  // dim1[31:16] | tile_dim0
    g1[4] = (int)(tile_d1);                                              // tile_dim1 | tile_dim2=0
    g1[5] = (int)(uint32_t)(stride0 & 0xFFFFFFFFu);                      // dim0_stride[31:0]
    g1[6] = (int)(uint32_t)((stride0 >> 32) & 0xFFFFu);                  // dim0_stride[47:32]
    g1[7] = 0;
    i32x4 z4 = (i32x4){0, 0, 0, 0};
#if defined(__clang_major__) && (__clang_major__ >= 23)
    i32x8 z8 = (i32x8){0, 0, 0, 0, 0, 0, 0, 0};
    __builtin_amdgcn_tensor_load_to_lds(g0, g1, z4, z4, z8, 0);
#else
    __builtin_amdgcn_tensor_load_to_lds(g0, g1, z4, z4, 0);
#endif
}
#endif

// ---------- fp32 -> bf16 convert ----------
__global__ void f32_to_bf16_kernel(const float* __restrict__ in,
                                   uint16_t* __restrict__ out, int n) {
    int i = blockIdx.x * blockDim.x + threadIdx.x;
    if (i < n) out[i] = f2bf(in[i]);
}

// ---------- LayerNorm: one block per row (D=1024, 256 thr x 4 elems) ----------
__global__ __launch_bounds__(256) void layernorm_bf16_kernel(
    const float* __restrict__ X, const float* __restrict__ g,
    const float* __restrict__ bta, uint16_t* __restrict__ out) {
    const int row = blockIdx.x;
    const float* x = X + (size_t)row * D_;
    const int c0 = threadIdx.x * 4;
    float4 v = *(const float4*)&x[c0];
    float s  = v.x + v.y + v.z + v.w;
    float ss = v.x*v.x + v.y*v.y + v.z*v.z + v.w*v.w;
    for (int off = 16; off; off >>= 1) {
        s  += __shfl_xor(s,  off);
        ss += __shfl_xor(ss, off);
    }
    __shared__ float s1[8], s2[8];
    if ((threadIdx.x & 31) == 0) { s1[threadIdx.x >> 5] = s; s2[threadIdx.x >> 5] = ss; }
    __syncthreads();
    float ts = 0.f, tss = 0.f;
    #pragma unroll
    for (int i = 0; i < 8; i++) { ts += s1[i]; tss += s2[i]; }
    const float mu = ts * (1.0f / D_);
    const float var = tss * (1.0f / D_) - mu * mu;
    const float rs = rsqrtf(var + 1e-5f);
    uint16_t* o = out + (size_t)row * D_;
    o[c0+0] = f2bf((v.x - mu) * rs * g[c0+0] + bta[c0+0]);
    o[c0+1] = f2bf((v.y - mu) * rs * g[c0+1] + bta[c0+1]);
    o[c0+2] = f2bf((v.z - mu) * rs * g[c0+2] + bta[c0+2]);
    o[c0+3] = f2bf((v.w - mu) * rs * g[c0+3] + bta[c0+3]);
}

// ---------- RoPE on bf16 q,k in place ----------
__global__ void rope_kernel(uint16_t* __restrict__ q, uint16_t* __restrict__ k) {
    int idx = blockIdx.x * blockDim.x + threadIdx.x;
    if (idx >= B_ * S_ * H_ * (HD_ / 2)) return;
    const int i = idx & 31;
    const int h = (idx >> 5) & (H_ - 1);
    const int s = (idx >> 9) & (S_ - 1);
    const int b = idx >> 20;
    const float invf = __expf((float)i * (-2.0f / HD_) * 9.210340371976184f); // ln(1e4)
    float sn, cs;
    __sincosf((float)s * invf, &sn, &cs);
    const size_t base = (((size_t)b * S_ + s) * H_ + h) * HD_;
    float q1 = bf2f(q[base + i]), q2 = bf2f(q[base + i + 32]);
    q[base + i]      = f2bf(q1 * cs - q2 * sn);
    q[base + i + 32] = f2bf(q2 * cs + q1 * sn);
    float k1 = bf2f(k[base + i]), k2 = bf2f(k[base + i + 32]);
    k[base + i]      = f2bf(k1 * cs - k2 * sn);
    k[base + i + 32] = f2bf(k2 * cs + k1 * sn);
}

// ---------- WMMA GEMM: C[M,N] = act(A[M,K] @ B[N,K]^T + bias) + resid ----------
// Block tile 128x128, K-step 32, double-buffered LDS filled by TDM (wave 0),
// DMA for k-step i+1 overlaps the WMMAs of k-step i.
#define BMt 128
#define BNt 128
#define BKt 32
#define LPg 40   // LDS pitch (elems): 32 + 8 pad == TDM pad 4dw/16dw

template <bool BIAS, bool RESID, bool GELU, bool OUTBF>
__global__ __launch_bounds__(256) void gemm_bf16_kernel(
    const uint16_t* __restrict__ A, const uint16_t* __restrict__ Bw,
    float* __restrict__ outF, uint16_t* __restrict__ outB,
    const float* __restrict__ bias, const float* __restrict__ resid,
    int M, int N, int K) {
    __shared__ __align__(16) uint16_t As[2][BMt * LPg];
    __shared__ __align__(16) uint16_t Bs[2][BNt * LPg];
    const int tid  = threadIdx.x;
    const int lane = tid & 31, wave = tid >> 5;
    const int wm = wave & 3, wn = wave >> 2;
    const int l15 = lane & 15, hi8 = (lane >> 4) * 8, kh8 = hi8;
    const int bm0 = blockIdx.y * BMt, bn0 = blockIdx.x * BNt;

    f32x8 acc[2][4];
    #pragma unroll
    for (int mt = 0; mt < 2; mt++)
        #pragma unroll
        for (int nt = 0; nt < 4; nt++) acc[mt][nt] = (f32x8){};

    const int nk = K / BKt;
#if HAVE_TDM
    const uint32_t ldsA[2] = { (uint32_t)(size_t)&As[0][0], (uint32_t)(size_t)&As[1][0] };
    const uint32_t ldsB[2] = { (uint32_t)(size_t)&Bs[0][0], (uint32_t)(size_t)&Bs[1][0] };
    if (wave == 0) {
        tdm_load_tile_2d(ldsA[0], A  + (size_t)bm0 * K, K, M, BKt, BMt, K);
        tdm_load_tile_2d(ldsB[0], Bw + (size_t)bn0 * K, K, N, BKt, BNt, K);
        __builtin_amdgcn_s_wait_tensorcnt(0);
    }
    __syncthreads();
    for (int ki = 0; ki < nk; ki++) {
        const int cur = ki & 1, nxt = cur ^ 1;
        const bool more = (ki + 1) < nk;
        if (wave == 0 && more) {                 // async DMA overlaps compute
            const int k1 = (ki + 1) * BKt;
            tdm_load_tile_2d(ldsA[nxt], A  + (size_t)bm0 * K + k1, K, M, BKt, BMt, K);
            tdm_load_tile_2d(ldsB[nxt], Bw + (size_t)bn0 * K + k1, K, N, BKt, BNt, K);
        }
        bf16x16 af[2], bfr[4];
        #pragma unroll
        for (int mt = 0; mt < 2; mt++)
            af[mt] = load_frag_lds(&As[cur][(wm * 32 + mt * 16 + l15) * LPg], kh8);
        #pragma unroll
        for (int nt = 0; nt < 4; nt++)
            bfr[nt] = load_frag_lds(&Bs[cur][(wn * 64 + nt * 16 + l15) * LPg], kh8);
        #pragma unroll
        for (int mt = 0; mt < 2; mt++)
            #pragma unroll
            for (int nt = 0; nt < 4; nt++)
                acc[mt][nt] = wmma_bf16(af[mt], bfr[nt], acc[mt][nt]);
        if (wave == 0 && more) __builtin_amdgcn_s_wait_tensorcnt(0);
        __syncthreads();
    }
#else
    for (int ki = 0; ki < nk; ki++) {
        const int k0 = ki * BKt;
        __syncthreads();
        #pragma unroll
        for (int i = 0; i < 2; i++) {
            int lin = tid + i * 256;
            int r = lin >> 2, c = (lin & 3) * 8;
            *(uint4*)&As[0][r * LPg + c] = *(const uint4*)&A [(size_t)(bm0 + r) * K + k0 + c];
            *(uint4*)&Bs[0][r * LPg + c] = *(const uint4*)&Bw[(size_t)(bn0 + r) * K + k0 + c];
        }
        __syncthreads();
        bf16x16 af[2], bfr[4];
        #pragma unroll
        for (int mt = 0; mt < 2; mt++)
            af[mt] = load_frag_lds(&As[0][(wm * 32 + mt * 16 + l15) * LPg], kh8);
        #pragma unroll
        for (int nt = 0; nt < 4; nt++)
            bfr[nt] = load_frag_lds(&Bs[0][(wn * 64 + nt * 16 + l15) * LPg], kh8);
        #pragma unroll
        for (int mt = 0; mt < 2; mt++)
            #pragma unroll
            for (int nt = 0; nt < 4; nt++)
                acc[mt][nt] = wmma_bf16(af[mt], bfr[nt], acc[mt][nt]);
    }
#endif

    // Straight-line epilogue (template-specialized, no runtime flag branches).
    #pragma unroll
    for (int mt = 0; mt < 2; mt++) {
        #pragma unroll
        for (int nt = 0; nt < 4; nt++) {
            const int gcol = bn0 + wn * 64 + nt * 16 + l15;
            float bv = 0.0f;
            if constexpr (BIAS) bv = bias[gcol];
            #pragma unroll
            for (int r = 0; r < 8; r++) {
                const int grow = bm0 + wm * 32 + mt * 16 + r + hi8;
                float v = acc[mt][nt][r] + bv;
                if constexpr (GELU) v = 0.5f * v * (1.0f + erff(v * 0.7071067811865475f));
                if constexpr (RESID) v += resid[(size_t)grow * N + gcol];
                if constexpr (OUTBF) outB[(size_t)grow * N + gcol] = f2bf(v);
                else                 outF[(size_t)grow * N + gcol] = v;
            }
        }
    }
}

// ---------- Flash attention (causal), bf16 in/out, fp32 online softmax ----------
#define APt 72   // LDS pitch: 64 + 8 pad, 144B rows keep 16B alignment

__global__ __launch_bounds__(256) void flash_attn_kernel(
    const uint16_t* __restrict__ Q, const uint16_t* __restrict__ Kc,
    const uint16_t* __restrict__ V, uint16_t* __restrict__ O) {
    __shared__ __align__(16) uint16_t Qs[128 * APt];
    __shared__ __align__(16) uint16_t Ks[64 * APt];
    __shared__ __align__(16) uint16_t Vt[64 * APt];   // transposed: [hd][key]
    __shared__ __align__(16) uint16_t Ps[8 * 16 * APt];

    const int tid  = threadIdx.x, lane = tid & 31, wave = tid >> 5;
    const int l15 = lane & 15, hi8 = (lane >> 4) * 8, kh8 = hi8;
    const int bh = blockIdx.y, b = bh / H_, h = bh % H_;
    const int q0 = blockIdx.x * 128;
    const int rowstride = H_ * HD_;

    const size_t qb = (((size_t)b * S_ + q0) * H_ + h) * HD_;
    #pragma unroll
    for (int i = 0; i < 4; i++) {
        int lin = tid + i * 256;
        int r = lin >> 3, c = (lin & 7) * 8;
        *(uint4*)&Qs[r * APt + c] = *(const uint4*)&Q[qb + (size_t)r * rowstride + c];
    }
    __syncthreads();
    bf16x16 qa[2];
    qa[0] = load_frag_lds(&Qs[(wave * 16 + l15) * APt + 0],  kh8);
    qa[1] = load_frag_lds(&Qs[(wave * 16 + l15) * APt + 32], kh8);

    f32x8 o[4];
    #pragma unroll
    for (int nt = 0; nt < 4; nt++) o[nt] = (f32x8){};
    float m_i[8], l_i[8];
    #pragma unroll
    for (int r = 0; r < 8; r++) { m_i[r] = -__builtin_inff(); l_i[r] = 0.f; }
    const int qrow_g = q0 + wave * 16;
    const int nkt = blockIdx.x * 2 + 2;   // causal: key tiles with kbase <= q0+127

    for (int kt = 0; kt < nkt; kt++) {
        const int k0 = kt * 64;
        __syncthreads();
        const size_t kb = (((size_t)b * S_ + k0) * H_ + h) * HD_;
        #pragma unroll
        for (int i = 0; i < 2; i++) {                 // K tile 64x64 row-major
            int lin = tid + i * 256;
            int r = lin >> 3, c = (lin & 7) * 8;
            *(uint4*)&Ks[r * APt + c] = *(const uint4*)&Kc[kb + (size_t)r * rowstride + c];
        }
        #pragma unroll
        for (int i = 0; i < 2; i++) {                 // V tile transposed -> Vt[hd][key]
            int lin = tid + i * 256;
            int r = lin >> 3, c8 = lin & 7;
            uint4 d = *(const uint4*)&V[kb + (size_t)r * rowstride + c8 * 8];
            const uint16_t* e = (const uint16_t*)&d;
            #pragma unroll
            for (int j = 0; j < 8; j++) Vt[(c8 * 8 + j) * APt + r] = e[j];
        }
        __syncthreads();

        // S = Q K^T
        f32x8 sc[4];
        #pragma unroll
        for (int st = 0; st < 4; st++) {
            f32x8 a = (f32x8){};
            #pragma unroll
            for (int ks = 0; ks < 2; ks++) {
                bf16x16 kf = load_frag_lds(&Ks[(st * 16 + l15) * APt + ks * 32], kh8);
                a = wmma_bf16(qa[ks], kf, a);
            }
            sc[st] = a;
        }

        float sv[4][8];
        #pragma unroll
        for (int st = 0; st < 4; st++)
            #pragma unroll
            for (int r = 0; r < 8; r++) {
                float xv = sc[st][r] * 0.125f;        // 1/sqrt(64)
                const int col = k0 + st * 16 + l15;
                const int row = qrow_g + r + hi8;
                sv[st][r] = (col > row) ? -__builtin_inff() : xv;
            }

        // online softmax; lanes 0-15 / 16-31 hold independent rows -> xor<16 reduce
        #pragma unroll
        for (int r = 0; r < 8; r++) {
            float rm = fmaxf(fmaxf(sv[0][r], sv[1][r]), fmaxf(sv[2][r], sv[3][r]));
            #pragma unroll
            for (int off = 1; off < 16; off <<= 1) rm = fmaxf(rm, __shfl_xor(rm, off));
            const float mold = m_i[r];
            const float mnew = fmaxf(mold, rm);
            const float meff = (mnew == -__builtin_inff()) ? 0.f : mnew;
            const float alpha = __expf(((mold == -__builtin_inff()) ? meff : mold) - meff);
            float rsum = 0.f;
            #pragma unroll
            for (int st = 0; st < 4; st++) {
                float p = __expf(sv[st][r] - meff);
                sv[st][r] = p; rsum += p;
            }
            #pragma unroll
            for (int off = 1; off < 16; off <<= 1) rsum += __shfl_xor(rsum, off);
            l_i[r] = l_i[r] * alpha + rsum;
            m_i[r] = mnew;
            #pragma unroll
            for (int nt = 0; nt < 4; nt++) o[nt][r] *= alpha;
        }

        // transpose P (C layout) -> A layout via wave-private LDS
        uint16_t* Pw = &Ps[wave * 16 * APt];
        #pragma unroll
        for (int st = 0; st < 4; st++)
            #pragma unroll
            for (int r = 0; r < 8; r++)
                Pw[(r + hi8) * APt + st * 16 + l15] = f2bf(sv[st][r]);
        __syncthreads();

        // O += P V
        #pragma unroll
        for (int ks = 0; ks < 2; ks++) {
            bf16x16 pf = load_frag_lds(&Pw[l15 * APt + ks * 32], kh8);
            #pragma unroll
            for (int nt = 0; nt < 4; nt++) {
                bf16x16 vf = load_frag_lds(&Vt[(nt * 16 + l15) * APt + ks * 32], kh8);
                o[nt] = wmma_bf16(pf, vf, o[nt]);
            }
        }
    }

    #pragma unroll
    for (int r = 0; r < 8; r++) {
        const float inv = l_i[r] > 0.f ? 1.0f / l_i[r] : 0.f;
        const int srow = qrow_g + r + hi8;
        const size_t ob = (((size_t)b * S_ + srow) * H_ + h) * HD_;
        #pragma unroll
        for (int nt = 0; nt < 4; nt++)
            O[ob + nt * 16 + l15] = f2bf(o[nt][r] * inv);
    }
}

// ---------------------------------------------------------------------------
extern "C" void kernel_launch(void* const* d_in, const int* in_sizes, int n_in,
                              void* d_out, int out_size, void* d_ws, size_t ws_size,
                              hipStream_t stream) {
    const float* x    = (const float*)d_in[0];
    /* d_in[1] = mask (bool) — causal mask applied analytically */
    const float* wq   = (const float*)d_in[2];
    const float* wk   = (const float*)d_in[3];
    const float* wv   = (const float*)d_in[4];
    const float* wo   = (const float*)d_in[5];
    const float* fc1w = (const float*)d_in[6];
    const float* fc1b = (const float*)d_in[7];
    const float* fc2w = (const float*)d_in[8];
    const float* fc2b = (const float*)d_in[9];
    const float* g1   = (const float*)d_in[10];
    const float* b1   = (const float*)d_in[11];
    const float* g2   = (const float*)d_in[12];
    const float* b2   = (const float*)d_in[13];

    char* ws = (char*)d_ws;
    size_t off = 0;
    auto alloc = [&](size_t bytes) -> char* {
        char* p = ws + off;
        off += (bytes + 255) & ~(size_t)255;
        return p;
    };
    uint16_t* wq_b   = (uint16_t*)alloc((size_t)D_ * D_ * 2);
    uint16_t* wk_b   = (uint16_t*)alloc((size_t)D_ * D_ * 2);
    uint16_t* wv_b   = (uint16_t*)alloc((size_t)D_ * D_ * 2);
    uint16_t* wo_b   = (uint16_t*)alloc((size_t)D_ * D_ * 2);
    uint16_t* fc1w_b = (uint16_t*)alloc((size_t)FF_ * D_ * 2);
    uint16_t* fc2w_b = (uint16_t*)alloc((size_t)D_ * FF_ * 2);
    uint16_t* h_b    = (uint16_t*)alloc((size_t)BS_ * D_ * 2);
    uint16_t* q_b    = (uint16_t*)alloc((size_t)BS_ * D_ * 2);
    uint16_t* k_b    = (uint16_t*)alloc((size_t)BS_ * D_ * 2);
    uint16_t* v_b    = (uint16_t*)alloc((size_t)BS_ * D_ * 2);
    uint16_t* attn_b = (uint16_t*)alloc((size_t)BS_ * D_ * 2);
    float*    x1     = (float*)   alloc((size_t)BS_ * D_ * 4);
    uint16_t* h2_b   = (uint16_t*)alloc((size_t)BS_ * D_ * 2);
    uint16_t* f1_b   = (uint16_t*)alloc((size_t)BS_ * FF_ * 2);

    const int nd = D_ * D_, nf = FF_ * D_;
    f32_to_bf16_kernel<<<(nd + 255) / 256, 256, 0, stream>>>(wq,   wq_b,   nd);
    f32_to_bf16_kernel<<<(nd + 255) / 256, 256, 0, stream>>>(wk,   wk_b,   nd);
    f32_to_bf16_kernel<<<(nd + 255) / 256, 256, 0, stream>>>(wv,   wv_b,   nd);
    f32_to_bf16_kernel<<<(nd + 255) / 256, 256, 0, stream>>>(wo,   wo_b,   nd);
    f32_to_bf16_kernel<<<(nf + 255) / 256, 256, 0, stream>>>(fc1w, fc1w_b, nf);
    f32_to_bf16_kernel<<<(nf + 255) / 256, 256, 0, stream>>>(fc2w, fc2w_b, nf);

    layernorm_bf16_kernel<<<BS_, 256, 0, stream>>>(x, g1, b1, h_b);

    dim3 gD(D_ / BNt, BS_ / BMt);
    gemm_bf16_kernel<false,false,false,true><<<gD, 256, 0, stream>>>(
        h_b, wq_b, nullptr, q_b, nullptr, nullptr, BS_, D_, D_);
    gemm_bf16_kernel<false,false,false,true><<<gD, 256, 0, stream>>>(
        h_b, wk_b, nullptr, k_b, nullptr, nullptr, BS_, D_, D_);
    gemm_bf16_kernel<false,false,false,true><<<gD, 256, 0, stream>>>(
        h_b, wv_b, nullptr, v_b, nullptr, nullptr, BS_, D_, D_);

    const int nr = B_ * S_ * H_ * (HD_ / 2);
    rope_kernel<<<(nr + 255) / 256, 256, 0, stream>>>(q_b, k_b);

    flash_attn_kernel<<<dim3(S_ / 128, B_ * H_), 256, 0, stream>>>(q_b, k_b, v_b, attn_b);

    gemm_bf16_kernel<false,true,false,false><<<gD, 256, 0, stream>>>(
        attn_b, wo_b, x1, nullptr, nullptr, x, BS_, D_, D_);

    layernorm_bf16_kernel<<<BS_, 256, 0, stream>>>(x1, g2, b2, h2_b);

    dim3 gFF(FF_ / BNt, BS_ / BMt);
    gemm_bf16_kernel<true,false,true,true><<<gFF, 256, 0, stream>>>(
        h2_b, fc1w_b, nullptr, f1_b, fc1b, nullptr, BS_, FF_, D_);
    gemm_bf16_kernel<true,true,false,false><<<gD, 256, 0, stream>>>(
        f1_b, fc2w_b, (float*)d_out, nullptr, fc2b, x1, BS_, D_, FF_);
}
